// DistWeightedLoss_78623671321304
// MI455X (gfx1250) — matrix operand order repaired and checked by reading, not compiled, not run on hardware
//
#include <hip/hip_runtime.h>
#include <hip/hip_bf16.h>
#include <cstdint>
#include <cstddef>

#define N_PTS   4096
#define D_DIM   128
#define M_INST  8
#define ALPHA_C 50.0f
#define BETA_C  2.0f
#define BASE_C  0.5f

typedef __attribute__((ext_vector_type(16))) _Float16 v16h;
typedef __attribute__((ext_vector_type(8)))  _Float16 v8h;
typedef __attribute__((ext_vector_type(4)))  _Float16 v4h;
typedef __attribute__((ext_vector_type(8)))  float    v8f;

// ---------------------------------------------------------------------------
// Stage 1: f32 -> f16 conversion of the embeddings (2 MB -> 1 MB, L2-resident)
// ---------------------------------------------------------------------------
__global__ void __launch_bounds__(256) convert_f16(const float* __restrict__ x,
                                                   _Float16* __restrict__ xh, int n4) {
    int i = blockIdx.x * blockDim.x + threadIdx.x;
    if (i < n4) {
        float4 v = ((const float4*)x)[i];
        v4h h;
        h[0] = (_Float16)v.x; h[1] = (_Float16)v.y;
        h[2] = (_Float16)v.z; h[3] = (_Float16)v.w;
        ((v4h*)xh)[i] = h;
    }
}

// ---------------------------------------------------------------------------
// Stage 2: sim = X * X^T via v_wmma_f32_16x16x32_f16.
// One wave computes a 16x64 strip: 4 N-tiles sharing one A fragment per
// K-step -> 16 WMMAs per wave, ~5 KB L2 traffic per 16x16 tile.
// Block = 256 threads = 8 waves -> 16 x 512 strip per block.
//
// A layout (16x32 f16): lane L holds row M=L&15;
//   lanes 0-15:  a[0..15]=K0..15 (contiguous per lane, same as B)
//   lanes 16-31: a[0..15]=K16..31
// B layout (32x16 f16): lane L holds column N=L&15;
//   lanes 0-15 hold K=0..15, lanes 16-31 hold K=16..31 (contiguous per lane).
// C/D layout (16x16 f32, 8 VGPRs): d[v] -> row M = v + 8*(L>=16), col N = L&15.
// ---------------------------------------------------------------------------
__global__ void __launch_bounds__(256) gram_wmma(const _Float16* __restrict__ xh,
                                                 float* __restrict__ sim) {
    const int wave = threadIdx.x >> 5;
    const int lane = threadIdx.x & 31;
    const int half = lane >> 4;
    const int lm   = lane & 15;
    const int i0 = blockIdx.x * 16;
    const int j0 = blockIdx.y * 512 + wave * 64;

    const _Float16* arow = xh + (size_t)(i0 + lm) * D_DIM;  // A: row i0+lm
    const _Float16* brow = xh + (size_t)(j0 + lm) * D_DIM;  // B tile 0 base

    v8f c0 = {}, c1 = {}, c2 = {}, c3 = {};
#pragma unroll
    for (int kk = 0; kk < D_DIM; kk += 32) {
        v8h alo = *(const v8h*)(arow + kk + half * 16);      // K = kk+half*16 .. +7
        v8h ahi = *(const v8h*)(arow + kk + half * 16 + 8);  // K = kk+half*16+8 ..
        v16h a;
#pragma unroll
        for (int t = 0; t < 8; ++t) { a[t] = alo[t]; a[t + 8] = ahi[t]; }
        const _Float16* bk = brow + kk + half * 16;          // K = kk+half*16 .. +15
        v16h b0 = *(const v16h*)(bk);
        v16h b1 = *(const v16h*)(bk + 16 * D_DIM);
        v16h b2 = *(const v16h*)(bk + 32 * D_DIM);
        v16h b3 = *(const v16h*)(bk + 48 * D_DIM);
        c0 = __builtin_amdgcn_wmma_f32_16x16x32_f16(false, a, false, b0, (short)0, c0, false, false);
        c1 = __builtin_amdgcn_wmma_f32_16x16x32_f16(false, a, false, b1, (short)0, c1, false, false);
        c2 = __builtin_amdgcn_wmma_f32_16x16x32_f16(false, a, false, b2, (short)0, c2, false, false);
        c3 = __builtin_amdgcn_wmma_f32_16x16x32_f16(false, a, false, b3, (short)0, c3, false, false);
    }
    float* srow = sim + (size_t)(i0 + 8 * half) * N_PTS + j0 + lm;
#pragma unroll
    for (int v = 0; v < 8; ++v) {
        srow[(size_t)v * N_PTS +  0] = c0[v];
        srow[(size_t)v * N_PTS + 16] = c1[v];
        srow[(size_t)v * N_PTS + 32] = c2[v];
        srow[(size_t)v * N_PTS + 48] = c3[v];
    }
}

// ---------------------------------------------------------------------------
// Stage 3: per-row pass. One block per row; row staged in LDS (16 KB) plus a
// weight array (16 KB). Computes neg mean/var, Gumbel-perturbed selection
// weights, deterministic bisection for the k-th largest threshold, then both
// log-sum-exp losses. All reductions are fixed-order (deterministic).
// ---------------------------------------------------------------------------
__device__ __forceinline__ float hash_u01(uint32_t x) {
    x ^= x >> 17; x *= 0xED5AD4BBu;
    x ^= x >> 11; x *= 0xAC4C1B51u;
    x ^= x >> 15; x *= 0x31848BABu;
    x ^= x >> 14;
    return (float)(x >> 8) * (1.0f / 16777216.0f);
}

__global__ void __launch_bounds__(256) row_pass(const float* __restrict__ sim,
                                                float* __restrict__ row_loss,
                                                float* __restrict__ row_pos,
                                                float* __restrict__ row_neg) {
    __shared__ __align__(16) float s[N_PTS];
    __shared__ float w[N_PTS];
    __shared__ float red[256];

    const int i   = blockIdx.x;
    const int t   = threadIdx.x;
    const int cls = i >> 3;  // targets = arange(N) // 8
    const float4* row4 = (const float4*)(sim + (size_t)i * N_PTS);

    for (int j = t; j < N_PTS / 4; j += 256) ((float4*)s)[j] = row4[j];
    __syncthreads();

    // ---- negative stats (sum, sumsq) ----
    float sum = 0.f, sq = 0.f;
    for (int j = t; j < N_PTS; j += 256)
        if ((j >> 3) != cls) { float v = s[j]; sum += v; sq += v * v; }
    red[t] = sum; __syncthreads();
    for (int o = 128; o > 0; o >>= 1) { if (t < o) red[t] += red[t + o]; __syncthreads(); }
    const float nsum = red[0]; __syncthreads();
    red[t] = sq; __syncthreads();
    for (int o = 128; o > 0; o >>= 1) { if (t < o) red[t] += red[t + o]; __syncthreads(); }
    const float nsq = red[0]; __syncthreads();

    const float cntn  = (float)(N_PTS - M_INST);
    const float mean  = nsum / cntn;
    const float var   = fmaxf(nsq / cntn - mean * mean, 1e-12f);  // population var
    const float inv2v = 0.5f / var;

    // ---- Gumbel-perturbed selection weights ----
    float wmax = -3.0e38f, wmin = 3.0e38f;
    for (int j = t; j < N_PTS; j += 256) {
        float wv;
        if ((j >> 3) != cls) {
            float z = s[j] - mean; z = z * z * inv2v;             // log-weight
            float u = hash_u01((uint32_t)(i * N_PTS + j) * 2654435761u + 0x9E3779B9u);
            float g = -logf(-logf(u + 1e-20f) + 1e-20f);          // Gumbel(0,1)
            wv = z + g;
            wmax = fmaxf(wmax, wv); wmin = fminf(wmin, wv);
        } else {
            wv = -3.0e38f;  // positives / diagonal never selected
        }
        w[j] = wv;
    }
    red[t] = wmax; __syncthreads();
    for (int o = 128; o > 0; o >>= 1) { if (t < o) red[t] = fmaxf(red[t], red[t + o]); __syncthreads(); }
    wmax = red[0]; __syncthreads();
    red[t] = wmin; __syncthreads();
    for (int o = 128; o > 0; o >>= 1) { if (t < o) red[t] = fminf(red[t], red[t + o]); __syncthreads(); }
    wmin = red[0]; __syncthreads();

    // ---- deterministic bisection for the k-th largest weight (k = 408) ----
    const int K = (N_PTS - M_INST) / 10;
    float lo = wmin - 1.0f, hi = wmax;
    for (int it = 0; it < 32; ++it) {
        const float mid = 0.5f * (lo + hi);
        float c = 0.f;
        for (int j = t; j < N_PTS; j += 256) if (w[j] > mid) c += 1.f;
        red[t] = c; __syncthreads();
        for (int o = 128; o > 0; o >>= 1) { if (t < o) red[t] += red[t + o]; __syncthreads(); }
        const float cnt = red[0]; __syncthreads();
        if (cnt >= (float)K) lo = mid; else hi = mid;   // keep count(w>lo) >= K
    }

    // ---- neg loss over selected negatives ----
    float se = 0.f;
    for (int j = t; j < N_PTS; j += 256)
        if (w[j] > lo) se += expf(ALPHA_C * (s[j] - BASE_C));
    red[t] = se; __syncthreads();
    for (int o = 128; o > 0; o >>= 1) { if (t < o) red[t] += red[t + o]; __syncthreads(); }
    const float negexp = red[0]; __syncthreads();

    if (t == 0) {
        float psum = 0.f, pexp = 0.f;
        const int p0 = cls * M_INST;
        for (int j = p0; j < p0 + M_INST; ++j)
            if (j != i) { float v = s[j]; psum += v; pexp += expf(-BETA_C * (v - BASE_C)); }
        const float pl = (2.0f / BETA_C) * log1pf(pexp);
        const float nl = (2.0f / ALPHA_C) * log1pf(negexp);
        row_loss[i] = pl + nl;
        row_pos[i]  = psum;
        row_neg[i]  = nsum;
    }
}

// ---------------------------------------------------------------------------
// Stage 4: deterministic fixed-order final reduction -> 4 scalars
// ---------------------------------------------------------------------------
__global__ void __launch_bounds__(256) finalize(const float* __restrict__ row_loss,
                                                const float* __restrict__ row_pos,
                                                const float* __restrict__ row_neg,
                                                float* __restrict__ out) {
    __shared__ float red[256];
    const int t = threadIdx.x;
    float a = 0.f, b = 0.f, c = 0.f;
    for (int j = t; j < N_PTS; j += 256) { a += row_loss[j]; b += row_pos[j]; c += row_neg[j]; }
    red[t] = a; __syncthreads();
    for (int o = 128; o > 0; o >>= 1) { if (t < o) red[t] += red[t + o]; __syncthreads(); }
    a = red[0]; __syncthreads();
    red[t] = b; __syncthreads();
    for (int o = 128; o > 0; o >>= 1) { if (t < o) red[t] += red[t + o]; __syncthreads(); }
    b = red[0]; __syncthreads();
    red[t] = c; __syncthreads();
    for (int o = 128; o > 0; o >>= 1) { if (t < o) red[t] += red[t + o]; __syncthreads(); }
    c = red[0]; __syncthreads();
    if (t == 0) {
        out[0] = a / (float)N_PTS;                                     // loss
        out[1] = 0.0f;                                                 // prec
        out[2] = b / (float)(N_PTS * (M_INST - 1));                    // pos_d
        out[3] = c / ((float)N_PTS * (float)(N_PTS - M_INST));         // neg_d
    }
}

// ---------------------------------------------------------------------------
// Launch: convert -> WMMA Gram -> per-row pass -> final reduce, all on stream.
// Workspace: [0,1MB) f16 X | [1MB,65MB) f32 sim | +48KB per-row results.
// ---------------------------------------------------------------------------
extern "C" void kernel_launch(void* const* d_in, const int* in_sizes, int n_in,
                              void* d_out, int out_size, void* d_ws, size_t ws_size,
                              hipStream_t stream) {
    (void)in_sizes; (void)n_in; (void)out_size; (void)ws_size;
    const float* x = (const float*)d_in[0];

    char* ws = (char*)d_ws;
    _Float16* xh  = (_Float16*)ws;                                    // 1 MB
    float*    sim = (float*)(ws + (size_t)(1 << 20));                 // 64 MB
    float*    rl  = (float*)(ws + (size_t)(1 << 20)
                                 + (size_t)N_PTS * N_PTS * sizeof(float));
    float*    rp  = rl + N_PTS;
    float*    rn  = rp + N_PTS;
    float*    out = (float*)d_out;

    convert_f16<<<dim3((N_PTS * D_DIM / 4 + 255) / 256), 256, 0, stream>>>(x, xh, N_PTS * D_DIM / 4);
    gram_wmma  <<<dim3(N_PTS / 16, N_PTS / 512), 256, 0, stream>>>(xh, sim);
    row_pass   <<<dim3(N_PTS), 256, 0, stream>>>(sim, rl, rp, rn);
    finalize   <<<dim3(1), 256, 0, stream>>>(rl, rp, rn, out);
}
